// KeyValueNetwork_54975581388885
// MI455X (gfx1250) — matrix operand order repaired and checked by reading, not compiled, not run on hardware
//
#include <hip/hip_runtime.h>

typedef __attribute__((ext_vector_type(16))) _Float16 v16h;
typedef __attribute__((ext_vector_type(8)))  _Float16 v8h;
typedef __attribute__((ext_vector_type(8)))  float    v8f;

#define D_ 512
#define C_ 1024
#define STEEP 10.0f

union AFrag { v16h v; struct { v8h lo; v8h hi; } s; };

// ---------------------------------------------------------------------------
// Prep 1: L2-normalize key rows, store f16 row-major (== column-major B for GEMM1)
// ---------------------------------------------------------------------------
__global__ __launch_bounds__(256)
void knorm_kernel(const float* __restrict__ key, _Float16* __restrict__ kh) {
    __shared__ float red[256];
    const int c = blockIdx.x;
    const float* row = key + (size_t)c * D_;
    float v0 = row[threadIdx.x];
    float v1 = row[threadIdx.x + 256];
    red[threadIdx.x] = v0 * v0 + v1 * v1;
    __syncthreads();
    for (int s = 128; s > 0; s >>= 1) {
        if ((int)threadIdx.x < s) red[threadIdx.x] += red[threadIdx.x + s];
        __syncthreads();
    }
    float inv = rsqrtf(red[0] + 1e-12f);
    kh[(size_t)c * D_ + threadIdx.x]       = (_Float16)(v0 * inv);
    kh[(size_t)c * D_ + threadIdx.x + 256] = (_Float16)(v1 * inv);
}

// ---------------------------------------------------------------------------
// Prep 2: vt[n][k] = (f16) val[k][n]   (column-major B operand for GEMM2)
// ---------------------------------------------------------------------------
__global__ __launch_bounds__(256)
void vtrans_kernel(const float* __restrict__ val, _Float16* __restrict__ vt) {
    __shared__ float tile[32][33];
    const int n0 = blockIdx.x * 32, k0 = blockIdx.y * 32;
    const int tx = threadIdx.x & 31, ty = threadIdx.x >> 5;  // ty: 0..7
    #pragma unroll
    for (int i = 0; i < 32; i += 8)
        tile[ty + i][tx] = val[(size_t)(k0 + ty + i) * C_ + (n0 + tx)];
    __syncthreads();
    #pragma unroll
    for (int i = 0; i < 32; i += 8)
        vt[(size_t)(n0 + ty + i) * C_ + (k0 + tx)] = (_Float16)tile[tx][ty + i];
}

// ---------------------------------------------------------------------------
// Main fused kernel: 32 query rows per block, 8 waves (wave32), 2 WMMA GEMMs.
// B fragments loaded once and shared by both 16-row tiles (mt pairing).
// ---------------------------------------------------------------------------
__global__ __launch_bounds__(256)
void kv_main_kernel(const float* __restrict__ q, const _Float16* __restrict__ kh,
                    const _Float16* __restrict__ vt, float* __restrict__ out) {
    __shared__ _Float16 sQ[32 * D_];   // 32 KB: normalized query tile, f16
    __shared__ _Float16 sW[32 * C_];   // 64 KB: sharpened weights, f16
    __shared__ float    sRed[256];
    __shared__ float    sInv[32];

    const int tid  = threadIdx.x;
    const int row0 = blockIdx.x * 32;
    const int lane = tid & 31;
    const int wv   = tid >> 5;     // wave 0..7 -> column strip wv*128
    const int lr   = lane & 15;
    const int lhi  = lane >> 4;

    // ---- Stage 0: coalesced load -> f16 LDS, then L2-normalize rows ----
    {
        // 32 rows x 512 f32 = 4096 float4; 256 threads x 16 iters, lane-contiguous
        const float4* qv = (const float4*)(q + (size_t)row0 * D_);
        #pragma unroll
        for (int i = 0; i < 16; ++i) {
            float4 v = qv[(size_t)i * 256 + tid];
            _Float16* dst = sQ + ((size_t)i * 256 + tid) * 4;
            dst[0] = (_Float16)v.x; dst[1] = (_Float16)v.y;
            dst[2] = (_Float16)v.z; dst[3] = (_Float16)v.w;
        }
        __syncthreads();

        const int r = tid >> 3, ch = tid & 7;     // 8 threads per row
        const v8h* wr = (const v8h*)(sQ + r * D_ + ch * 64);
        float ss = 0.f;
        #pragma unroll
        for (int i = 0; i < 8; ++i) {
            v8h v = wr[i];
            #pragma unroll
            for (int j = 0; j < 8; ++j) { float f = (float)v[j]; ss += f * f; }
        }
        sRed[tid] = ss;
        __syncthreads();
        if (tid < 32) {
            float s = 0.f;
            #pragma unroll
            for (int j = 0; j < 8; ++j) s += sRed[tid * 8 + j];
            sInv[tid] = rsqrtf(s + 1e-12f);
        }
        __syncthreads();
        const float inv = sInv[r];
        _Float16* dq = sQ + r * D_ + ch * 64;
        #pragma unroll
        for (int i = 0; i < 64; ++i) dq[i] = (_Float16)((float)dq[i] * inv);
        __syncthreads();
    }

    // ---- Stage 1: sim = qn @ knT via WMMA (B shared by both row tiles),
    //      softabs -> sW (f16) ----
    {
        const _Float16* aBase0 = sQ + lr * D_ + lhi * 16;         // rows 0..15
        const _Float16* aBase1 = aBase0 + 16 * D_;                // rows 16..31
        for (int nt = 0; nt < 8; ++nt) {
            const int col = wv * 128 + nt * 16 + lr;
            const _Float16* bBase = kh + (size_t)col * D_ + lhi * 16;
            v8f acc0 = {}, acc1 = {};
            #pragma unroll
            for (int k = 0; k < D_; k += 32) {
                v16h b = *(const v16h*)(bBase + k);
                AFrag a0, a1;
                a0.s.lo = *(const v8h*)(aBase0 + k);
                a0.s.hi = *(const v8h*)(aBase0 + k + 8);
                a1.s.lo = *(const v8h*)(aBase1 + k);
                a1.s.hi = *(const v8h*)(aBase1 + k + 8);
                acc0 = __builtin_amdgcn_wmma_f32_16x16x32_f16(
                        false, a0.v, false, b, (short)0, acc0, false, false);
                acc1 = __builtin_amdgcn_wmma_f32_16x16x32_f16(
                        false, a1.v, false, b, (short)0, acc1, false, false);
            }
            #pragma unroll
            for (int i = 0; i < 8; ++i) {
                const int m = lhi * 8 + i;
                float x0 = acc0[i];
                float x1 = acc1[i];
                float s0 = 1.f / (1.f + __expf(-(STEEP * x0 - 5.f)))
                         + 1.f / (1.f + __expf( STEEP * x0 + 5.f));
                float s1 = 1.f / (1.f + __expf(-(STEEP * x1 - 5.f)))
                         + 1.f / (1.f + __expf( STEEP * x1 + 5.f));
                sW[m * C_ + col]        = (_Float16)s0;
                sW[(m + 16) * C_ + col] = (_Float16)s1;
            }
        }
    }
    __syncthreads();

    // ---- Stage 2: row sums of sharp -> 1/sum per row ----
    {
        const int r = tid >> 3, ch = tid & 7;
        const v8h* wr = (const v8h*)(sW + r * C_ + ch * 128);
        float s = 0.f;
        #pragma unroll
        for (int i = 0; i < 16; ++i) {
            v8h v = wr[i];
            #pragma unroll
            for (int j = 0; j < 8; ++j) s += (float)v[j];
        }
        sRed[tid] = s;
        __syncthreads();
        if (tid < 32) {
            float t = 0.f;
            #pragma unroll
            for (int j = 0; j < 8; ++j) t += sRed[tid * 8 + j];
            sInv[tid] = 1.0f / t;
        }
        __syncthreads();
    }

    // ---- Stage 3: out = (sharp @ val) * (1/rowsum) via WMMA,
    //      B shared by both row tiles ----
    {
        const _Float16* aBase0 = sW + lr * C_ + lhi * 16;         // rows 0..15
        const _Float16* aBase1 = aBase0 + 16 * C_;                // rows 16..31
        float scale0[8], scale1[8];
        #pragma unroll
        for (int i = 0; i < 8; ++i) {
            scale0[i] = sInv[lhi * 8 + i];
            scale1[i] = sInv[16 + lhi * 8 + i];
        }
        for (int nt = 0; nt < 8; ++nt) {
            const int col = wv * 128 + nt * 16 + lr;
            const _Float16* bBase = vt + (size_t)col * C_ + lhi * 16;
            v8f acc0 = {}, acc1 = {};
            #pragma unroll 8
            for (int k = 0; k < C_; k += 32) {
                v16h b = *(const v16h*)(bBase + k);
                AFrag a0, a1;
                a0.s.lo = *(const v8h*)(aBase0 + k);
                a0.s.hi = *(const v8h*)(aBase0 + k + 8);
                a1.s.lo = *(const v8h*)(aBase1 + k);
                a1.s.hi = *(const v8h*)(aBase1 + k + 8);
                acc0 = __builtin_amdgcn_wmma_f32_16x16x32_f16(
                        false, a0.v, false, b, (short)0, acc0, false, false);
                acc1 = __builtin_amdgcn_wmma_f32_16x16x32_f16(
                        false, a1.v, false, b, (short)0, acc1, false, false);
            }
            #pragma unroll
            for (int i = 0; i < 8; ++i) {
                const int m = lhi * 8 + i;
                out[(size_t)(row0 + m) * C_ + col]      = acc0[i] * scale0[i];
                out[(size_t)(row0 + m + 16) * C_ + col] = acc1[i] * scale1[i];
            }
        }
    }
}

// ---------------------------------------------------------------------------
extern "C" void kernel_launch(void* const* d_in, const int* in_sizes, int n_in,
                              void* d_out, int out_size, void* d_ws, size_t ws_size,
                              hipStream_t stream) {
    (void)n_in; (void)out_size; (void)ws_size;
    const float* q   = (const float*)d_in[0];   // (B, D)
    const float* key = (const float*)d_in[1];   // (C, D)
    const float* val = (const float*)d_in[2];   // (C, C)
    float* out = (float*)d_out;                 // (B, C)

    const int B = in_sizes[0] / D_;

    _Float16* kh = (_Float16*)d_ws;                                   // 1 MB
    _Float16* vt = (_Float16*)((char*)d_ws + (size_t)C_ * D_ * 2);    // 2 MB

    knorm_kernel<<<C_, 256, 0, stream>>>(key, kh);
    vtrans_kernel<<<dim3(C_ / 32, C_ / 32), 256, 0, stream>>>(val, vt);
    kv_main_kernel<<<B / 32, 256, 0, stream>>>(q, kh, vt, out);
}
